// MultiHeadAttention_71287867179085
// MI455X (gfx1250) — matrix operand order repaired and checked by reading, not compiled, not run on hardware
//
#include <hip/hip_runtime.h>

// ---------------------------------------------------------------------------
// MHA forward for MI455X (gfx1250, wave32, WMMA).
// ~103 GFLOP over ~85 MB -> compute bound -> all GEMMs + QK^T / PV on
// v_wmma_f32_16x16x32_bf16 (f32 accumulate). Fragment-layout-aware loads
// (contiguous 32B per lane), fused v_max_num_f32_dpp row reductions,
// row-sums via ones-matrix WMMA, 64-key softmax steps, scale folded into Q.
// ---------------------------------------------------------------------------

typedef __attribute__((ext_vector_type(16))) __bf16 v16bf;
typedef __attribute__((ext_vector_type(8)))  __bf16 v8bf;
typedef __attribute__((ext_vector_type(8)))  float  v8f;

static constexpr int Bsz = 2, S = 2048, DM = 1024, H = 16, DH = 64;
static constexpr int Mrows = Bsz * S;   // 4096 rows for the projection GEMMs

#define DEV __device__ __forceinline__

DEV v16bf pack2(v8bf lo, v8bf hi) {
  v16bf r;
#pragma unroll
  for (int i = 0; i < 8; ++i) { r[i] = lo[i]; r[i + 8] = hi[i]; }
  return r;
}

DEV v8f wmma_bf16(v16bf a, v16bf b, v8f c) {
  // D = A(16x32 bf16) * B(32x16 bf16) + C(f32)
  return __builtin_amdgcn_wmma_f32_16x16x32_bf16(
      false, a, false, b, (short)0, c, false, false);
}

// 16-lane all-reduce max, one fused VALU-DPP instruction per step
// (x = max(x, dpp_perm(x))). Lanes 0-15 / 16-31 reduce independently,
// matching the WMMA C-fragment row groups.
DEV float rowmax16(float x) {
  asm("v_max_num_f32_dpp %0, %1, %1 quad_perm:[1,0,3,2] row_mask:0xf bank_mask:0xf bound_ctrl:1"
      : "=v"(x) : "v"(x));
  asm("v_max_num_f32_dpp %0, %1, %1 quad_perm:[2,3,0,1] row_mask:0xf bank_mask:0xf bound_ctrl:1"
      : "=v"(x) : "v"(x));
  asm("v_max_num_f32_dpp %0, %1, %1 row_half_mirror row_mask:0xf bank_mask:0xf bound_ctrl:1"
      : "=v"(x) : "v"(x));
  asm("v_max_num_f32_dpp %0, %1, %1 row_mirror row_mask:0xf bank_mask:0xf bound_ctrl:1"
      : "=v"(x) : "v"(x));
  return x;
}

// ---------------------------------------------------------------------------
// f32 -> bf16 weight conversion (grid-stride)
// ---------------------------------------------------------------------------
__global__ void __launch_bounds__(256) f32_to_bf16(const float* __restrict__ src,
                                                   __bf16* __restrict__ dst, int n) {
  int i = blockIdx.x * blockDim.x + threadIdx.x;
  const int stride = gridDim.x * blockDim.x;
  for (; i < n; i += stride) dst[i] = (__bf16)src[i];
}

// ---------------------------------------------------------------------------
// Y[m,n] = sum_k X[m,k] * W[n,k]   (torch Linear: y = x @ W^T)
//   M = 4096, N = K = 1024. One 16x64 output strip per wave: one A fragment
//   feeds 4 WMMAs per 32-wide K-step.
// OMODE: 0 -> bf16 [B,H,S,Dh] (Q/K), 1 -> bf16 [B,H,Dh,S] (V^T),
//        2 -> f32 row-major [M, DM] (final output).
// ---------------------------------------------------------------------------
template <bool A_F32, int OMODE>
__global__ void __launch_bounds__(256, 1)
gemm_xwT(const void* __restrict__ Xv, const __bf16* __restrict__ W,
         void* __restrict__ Yv) {
  const int wave = blockIdx.x * (blockDim.x >> 5) + (threadIdx.x >> 5);
  const int lane = threadIdx.x & 31;
  const int half = lane >> 4, l16 = lane & 15;
  const int NT = DM / 64;                 // 16 N-strips of 64
  const int mt = wave / NT, nt = wave % NT;
  const int mrow = mt * 16 + l16;
  const __bf16* wrow[4];
#pragma unroll
  for (int p = 0; p < 4; ++p)
    wrow[p] = W + (size_t)(nt * 64 + p * 16 + l16) * DM;

  v8f acc[4] = {{}, {}, {}, {}};
  for (int k0 = 0; k0 < DM; k0 += 32) {
    v16bf a;
    if (A_F32) {
      const float* pa = (const float*)Xv + (size_t)mrow * DM + k0;
      v8f lo = *(const v8f*)(pa + half * 16);
      v8f hi = *(const v8f*)(pa + half * 16 + 8);
#pragma unroll
      for (int i = 0; i < 8; ++i) { a[i] = (__bf16)lo[i]; a[i + 8] = (__bf16)hi[i]; }
      if (k0 + 32 < DM) __builtin_prefetch(pa + 32, 0, 0);   // global_prefetch_b8
    } else {
      const __bf16* pa = (const __bf16*)Xv + (size_t)mrow * DM + k0;
      a = pack2(*(const v8bf*)(pa + half * 16),
                *(const v8bf*)(pa + half * 16 + 8));
      if (k0 + 32 < DM) __builtin_prefetch(pa + 32, 0, 0);
    }
#pragma unroll
    for (int p = 0; p < 4; ++p) {
      v16bf b = *(const v16bf*)(wrow[p] + k0 + half * 16);
      acc[p] = wmma_bf16(a, b, acc[p]);
    }
  }

  // C/D layout: lane holds column l16; VGPR v holds row v + (half?8:0).
#pragma unroll
  for (int p = 0; p < 4; ++p) {
    const int gn = nt * 64 + p * 16 + l16;
#pragma unroll
    for (int v = 0; v < 8; ++v) {
      const int gm = mt * 16 + v + half * 8;
      if (OMODE == 2) {
        ((float*)Yv)[(size_t)gm * DM + gn] = acc[p][v];
      } else {
        const int bb = gm / S, ss = gm % S;
        const int hh = gn / DH, dd = gn % DH;
        const __bf16 val = (__bf16)acc[p][v];
        if (OMODE == 0)
          ((__bf16*)Yv)[(((size_t)(bb * H + hh)) * S + ss) * DH + dd] = val;
        else  // V stored transposed: [B,H,Dh,S]
          ((__bf16*)Yv)[(((size_t)(bb * H + hh)) * DH + dd) * S + ss] = val;
      }
    }
  }
}

// ---------------------------------------------------------------------------
// One 64-key flash-attention step: 4 score tiles (8 WMMAs), one softmax
// update, then P(16x64) x V(64x64) as 8 WMMAs + 2 ones-matrix WMMAs for the
// row sums. Q already carries the 1/sqrt(Dh) scale.
// ---------------------------------------------------------------------------
template <bool MASKED>
DEV void attn_step64(int k0, int qbase, int half, int l16,
                     const __bf16* __restrict__ Kb,
                     const __bf16* __restrict__ Vb,
                     __bf16* ldsPw, const v16bf* aq, v16bf bones,
                     float* mrun, v8f* o) {
  v8f s[4];
#pragma unroll
  for (int j = 0; j < 4; ++j) {
    v8f sj = {};
#pragma unroll
    for (int c = 0; c < 2; ++c) {
      const __bf16* kr = Kb + (size_t)(k0 + j * 16 + l16) * DH + c * 32 + half * 16;
      sj = wmma_bf16(aq[c], *(const v16bf*)kr, sj);
    }
    s[j] = sj;
  }
#pragma unroll
  for (int v = 0; v < 8; ++v) {
    float x[4];
#pragma unroll
    for (int j = 0; j < 4; ++j) {
      x[j] = s[j][v];
      if (MASKED) {
        const int m = qbase + v + half * 8;          // global query row
        if (k0 + j * 16 + l16 > m) x[j] = -1.0e30f;  // key = k0 + j*16 + l16
      }
    }
    float rmax = fmaxf(fmaxf(x[0], x[1]), fmaxf(x[2], x[3]));
    rmax = rowmax16(rmax);
    const float mnew = fmaxf(mrun[v], rmax);
    const float alpha = __expf(mrun[v] - mnew);
    mrun[v] = mnew;
#pragma unroll
    for (int t = 0; t < 5; ++t) o[t][v] *= alpha;    // O tiles + row-sum acc
    const int rbase = (v + half * 8) * 64;
#pragma unroll
    for (int j = 0; j < 4; ++j)
      ldsPw[rbase + j * 16 + l16] = (__bf16)__expf(x[j] - mnew);
  }
  asm volatile("s_wait_dscnt 0" ::: "memory");       // cross-lane LDS ordering
  const __bf16* pr = ldsPw + l16 * 64;
  v16bf ap0 = pack2(*(const v8bf*)(pr + half * 16),
                    *(const v8bf*)(pr + half * 16 + 8));
  v16bf ap1 = pack2(*(const v8bf*)(pr + 32 + half * 16),
                    *(const v8bf*)(pr + 32 + half * 16 + 8));
#pragma unroll
  for (int t = 0; t < 4; ++t) {
    const __bf16* vr = Vb + (size_t)(t * 16 + l16) * S;
    o[t] = wmma_bf16(ap0, *(const v16bf*)(vr + k0 + half * 16), o[t]);
    o[t] = wmma_bf16(ap1, *(const v16bf*)(vr + k0 + 32 + half * 16), o[t]);
  }
  o[4] = wmma_bf16(ap0, bones, o[4]);
  o[4] = wmma_bf16(ap1, bones, o[4]);
}

// ---------------------------------------------------------------------------
// Flash attention, causal. One wave per (b, h, 16-row q tile); 64 keys/step.
// Unmasked bulk loop + exactly one masked tail step (tail start k0 <= 1984,
// so all 64 loaded key rows stay inside S=2048).
// ---------------------------------------------------------------------------
__global__ void __launch_bounds__(128, 1)
flash_attn(const __bf16* __restrict__ Q, const __bf16* __restrict__ K,
           const __bf16* __restrict__ VT, __bf16* __restrict__ O) {
  __shared__ alignas(16) __bf16 ldsP[4][16 * 64];   // per-wave P staging (2KB)
  const int w    = threadIdx.x >> 5;
  const int lane = threadIdx.x & 31;
  const int half = lane >> 4, l16 = lane & 15;
  const int gw = blockIdx.x * 4 + w;
  const int qt = gw & 127;                 // q tile  (S/16 = 128)
  const int h  = (gw >> 7) & (H - 1);
  const int b  = gw >> 11;
  const size_t bh = (size_t)(b * H + h);
  const __bf16* Qb = Q  + bh * S * DH;
  const __bf16* Kb = K  + bh * S * DH;
  const __bf16* Vb = VT + bh * DH * S;     // [Dh, S]
  const int qbase = qt * 16;

  v16bf aq[2];
  {
    const __bf16* qr = Qb + (size_t)(qbase + l16) * DH;
#pragma unroll
    for (int c = 0; c < 2; ++c) {
      aq[c] = pack2(*(const v8bf*)(qr + c * 32 + half * 16),
                    *(const v8bf*)(qr + c * 32 + half * 16 + 8));
      // fold 1/sqrt(Dh) = 0.125 into Q (power of two: exact in bf16)
#pragma unroll
      for (int i = 0; i < 16; ++i)
        aq[c][i] = (__bf16)((float)aq[c][i] * 0.125f);
    }
  }
  v16bf bones;
#pragma unroll
  for (int i = 0; i < 16; ++i) bones[i] = (__bf16)1.0f;

  float mrun[8];
#pragma unroll
  for (int v = 0; v < 8; ++v) mrun[v] = -1.0e30f;
  v8f zero = {};
  v8f o[5] = {zero, zero, zero, zero, zero};  // 4 O tiles + row-sum tile

  int k0 = 0;
  for (; k0 + 64 <= qbase; k0 += 64)       // fully-valid steps: no masking
    attn_step64<false>(k0, qbase, half, l16, Kb, Vb, ldsP[w], aq, bones, mrun, o);
  // single masked step covers the remaining <=64 keys incl. the diagonal
  attn_step64<true>(k0, qbase, half, l16, Kb, Vb, ldsP[w], aq, bones, mrun, o);

  // epilogue: divide by row sums (replicated across all 16 lanes by the
  // ones-matrix WMMA), write O as bf16 [B*S, DM]
#pragma unroll
  for (int v = 0; v < 8; ++v) {
    const float inv = 1.0f / o[4][v];
    const int m = qbase + v + half * 8;
#pragma unroll
    for (int t = 0; t < 4; ++t)
      O[((size_t)(b * S + m)) * DM + h * DH + t * 16 + l16] =
          (__bf16)(o[t][v] * inv);
  }
}

// ---------------------------------------------------------------------------
// Launch: convert weights, 3 projection GEMMs, flash attention, output GEMM.
// Inputs: 0=query 1=key 2=value 3=freqs(unused) 4=mask(causal, hardcoded)
//         5=Wq 6=Wk 7=Wv 8=Wo.  Workspace: 4x2MB weights + 4x8MB tensors.
// ---------------------------------------------------------------------------
extern "C" void kernel_launch(void* const* d_in, const int* in_sizes, int n_in,
                              void* d_out, int out_size, void* d_ws,
                              size_t ws_size, hipStream_t stream) {
  (void)in_sizes; (void)n_in; (void)out_size; (void)ws_size;
  const float* q  = (const float*)d_in[0];
  const float* k  = (const float*)d_in[1];
  const float* v  = (const float*)d_in[2];
  const float* Wq = (const float*)d_in[5];
  const float* Wk = (const float*)d_in[6];
  const float* Wv = (const float*)d_in[7];
  const float* Wo = (const float*)d_in[8];

  char* ws = (char*)d_ws;
  const size_t WB = (size_t)DM * DM * sizeof(__bf16);          // 2 MB
  const size_t QB = (size_t)Bsz * H * S * DH * sizeof(__bf16); // 8 MB
  __bf16* Wq_b = (__bf16*)(ws);
  __bf16* Wk_b = (__bf16*)(ws + WB);
  __bf16* Wv_b = (__bf16*)(ws + 2 * WB);
  __bf16* Wo_b = (__bf16*)(ws + 3 * WB);
  __bf16* Qb   = (__bf16*)(ws + 4 * WB);
  __bf16* Kb   = (__bf16*)(ws + 4 * WB + QB);
  __bf16* Vt   = (__bf16*)(ws + 4 * WB + 2 * QB);
  __bf16* Ob   = (__bf16*)(ws + 4 * WB + 3 * QB);

  const int nW = DM * DM;
  f32_to_bf16<<<512, 256, 0, stream>>>(Wq, Wq_b, nW);
  f32_to_bf16<<<512, 256, 0, stream>>>(Wk, Wk_b, nW);
  f32_to_bf16<<<512, 256, 0, stream>>>(Wv, Wv_b, nW);
  f32_to_bf16<<<512, 256, 0, stream>>>(Wo, Wo_b, nW);

  const int gemmBlocks = (Mrows / 16) * (DM / 64) / 8;  // 8 waves/block = 512
  gemm_xwT<true, 0><<<gemmBlocks, 256, 0, stream>>>(q, Wq_b, Qb);
  gemm_xwT<true, 0><<<gemmBlocks, 256, 0, stream>>>(k, Wk_b, Kb);
  gemm_xwT<true, 1><<<gemmBlocks, 256, 0, stream>>>(v, Wv_b, Vt);

  flash_attn<<<Bsz * H * (S / 16) / 4, 128, 0, stream>>>(Qb, Kb, Vt, Ob);

  gemm_xwT<false, 2><<<gemmBlocks, 256, 0, stream>>>(Ob, Wo_b, d_out);
}
